// Temporal_SelfAttention_18133351923888
// MI455X (gfx1250) — compile-verified, matrix operand order checked by reading, not run on hardware
//
#include <hip/hip_runtime.h>

// ---------------------------------------------------------------------------
// Temporal self-attention for MI455X (gfx1250, wave32, WMMA bf16 path)
//   B=4, L=2048, D=256, H=8, dh=32
//   Pass 0: fp32 -> bf16 staging of x, Wq, Wk, Wv (one-shot, vectorized)
//   Pass 1: fused QKV projection (pure bf16 loads + v_wmma inner loop)
//   Pass 2: flash attention, 64-key blocks; online softmax where the row-sum
//           accumulator is itself a WMMA (P @ ones) -> no sum shuffles.
// ---------------------------------------------------------------------------

typedef __bf16 v16bf __attribute__((ext_vector_type(16)));
typedef __bf16 v8bf  __attribute__((ext_vector_type(8)));
typedef float  v8f   __attribute__((ext_vector_type(8)));
typedef float  v4f   __attribute__((ext_vector_type(4)));

constexpr int B_  = 4;
constexpr int L_  = 2048;
constexpr int D_  = 256;
constexpr int H_  = 8;
constexpr int DH  = 32;                 // head dim
constexpr int M_  = B_ * L_;            // 8192 token rows
constexpr float SCALE = 0.17677669529663687f;   // 1/sqrt(32)
constexpr int SBLK = 64;                // keys per flash-attention step
constexpr int PSTRIDE = 72;             // LDS row stride (bf16): 36 banks/row -> conflict-free

// 16x32 bf16 WMMA fragment (A or B operand) from bf16 memory/LDS.
// Lane holds row (lane&15); elements 0..7 = c..c+7, 8..15 = 16+c..16+c+7
// where c = 8*(lane>=16).  (Per documented 16-bit operand layout.)
__device__ inline v16bf frag_bf16(const __bf16* __restrict__ p, int c) {
  v8bf lo = *(const v8bf*)(p + c);
  v8bf hi = *(const v8bf*)(p + 16 + c);
  v16bf f;
#pragma unroll
  for (int i = 0; i < 8; ++i) {
    f[i]     = lo[i];
    f[8 + i] = hi[i];
  }
  return f;
}

// ---------------------------------------------------------------------------
// Pass 0: vectorized fp32 -> bf16 conversion (8 elements/thread).
// ---------------------------------------------------------------------------
__global__ __launch_bounds__(256)
void cvt_bf16_kernel(const float* __restrict__ src, __bf16* __restrict__ dst, int n)
{
  const int i = (blockIdx.x * 256 + threadIdx.x) * 8;
  if (i >= n) return;
  v4f a = *(const v4f*)(src + i);
  v4f b = *(const v4f*)(src + i + 4);
  v8bf o;
#pragma unroll
  for (int j = 0; j < 4; ++j) {
    o[j]     = (__bf16)a[j];
    o[4 + j] = (__bf16)b[j];
  }
  *(v8bf*)(dst + i) = o;
}

// ---------------------------------------------------------------------------
// Pass 1: fused QKV projection.
//   One wave -> one 16x16 tile of (q,k,v); A-fragment of x reused 3x.
//   Inner loop: 8 x (8 b128 bf16 loads + 3 v_wmma).
//   q,k written bf16 [B,H,L,32] (q pre-scaled by 1/sqrt(dh));
//   v written transposed [B,H,32,L] so P@V's B-operand is contiguous in keys.
// ---------------------------------------------------------------------------
__global__ __launch_bounds__(256)
void qkv_proj_kernel(const __bf16* __restrict__ xb,
                     const __bf16* __restrict__ wqb,
                     const __bf16* __restrict__ wkb,
                     const __bf16* __restrict__ wvb,
                     __bf16* __restrict__ qws,
                     __bf16* __restrict__ kws,
                     __bf16* __restrict__ vws)
{
  const int wave = threadIdx.x >> 5;
  const int lane = threadIdx.x & 31;
  const int lm   = lane & 15;
  const int c    = (lane >> 4) << 3;          // 0 or 8

  const int tile = blockIdx.x * 8 + wave;     // (M/16)*(D/16) = 8192 tiles
  const int mt   = tile >> 4;
  const int nt   = tile & 15;

  const __bf16* arow = xb  + (size_t)(mt * 16 + lm) * D_;
  const __bf16* qrow = wqb + (size_t)(nt * 16 + lm) * D_;  // rows of W = cols of W^T
  const __bf16* krow = wkb + (size_t)(nt * 16 + lm) * D_;
  const __bf16* vrow = wvb + (size_t)(nt * 16 + lm) * D_;

  v8f accq = {}, acck = {}, accv = {};
#pragma unroll 2
  for (int k0 = 0; k0 < D_; k0 += 32) {
    v16bf a  = frag_bf16(arow + k0, c);
    v16bf bq = frag_bf16(qrow + k0, c);
    v16bf bk = frag_bf16(krow + k0, c);
    v16bf bv = frag_bf16(vrow + k0, c);
    accq = __builtin_amdgcn_wmma_f32_16x16x32_bf16(false, a, false, bq, (short)0, accq, false, false);
    acck = __builtin_amdgcn_wmma_f32_16x16x32_bf16(false, a, false, bk, (short)0, acck, false, false);
    accv = __builtin_amdgcn_wmma_f32_16x16x32_bf16(false, a, false, bv, (short)0, accv, false, false);
  }

  // C/D layout: VGPR r, lane l -> (row = r + 8*(l>=16), col = l&15)
  const int n    = nt * 16 + lm;
  const int h    = n >> 5;
  const int d    = n & 31;
  const int bidx = (mt * 16) / L_;            // tiles never straddle batch (L%16==0)
  const int tok0 = (mt * 16) % L_ + ((lane >> 4) << 3);

  const size_t qkhead = ((size_t)(bidx * H_ + h) * L_) * DH + d;   // + tok*DH
  const size_t vhead  = ((size_t)(bidx * H_ + h) * DH + d) * L_;   // + tok
#pragma unroll
  for (int r = 0; r < 8; ++r) {
    const int tok = tok0 + r;
    qws[qkhead + (size_t)tok * DH] = (__bf16)(accq[r] * SCALE);
    kws[qkhead + (size_t)tok * DH] = (__bf16)acck[r];
    vws[vhead + tok]               = (__bf16)accv[r];
  }
}

// ---------------------------------------------------------------------------
// Pass 2: flash attention, 64 keys per step.
//   Block = 8 waves; wave owns 16 query rows (block covers 128 queries).
//   Per step: 4x WMMA (S), row-max via 4 shuffles/row, exp in fp32,
//   P repack C->A layout via per-wave LDS, then 6x WMMA:
//     o0/o1 += P @ V   (two 32-key sub-blocks x two dh halves)
//     o2    += P @ 1   (row-sum accumulator; every lane gets the sum)
//   The o2 trick removes all sum-reduction shuffles and keeps the
//   normalizer consistent with the bf16-rounded numerator.
// ---------------------------------------------------------------------------
__global__ __launch_bounds__(256)
void attn_kernel(const __bf16* __restrict__ qws,
                 const __bf16* __restrict__ kws,
                 const __bf16* __restrict__ vws,
                 float* __restrict__ out)
{
  __shared__ __bf16 pbuf[8][16 * PSTRIDE];

  const int wave = threadIdx.x >> 5;
  const int lane = threadIdx.x & 31;
  const int lm   = lane & 15;
  const int c    = (lane >> 4) << 3;          // 0 or 8
  const int hi8  = (lane >> 4) << 3;          // C-layout row offset

  const int nq   = L_ / 128;                  // 16 query blocks per (b,h)
  const int qblk = blockIdx.x % nq;
  const int bh   = blockIdx.x / nq;           // b*H + h

  const __bf16* qhead = qws + (size_t)bh * L_ * DH;
  const __bf16* khead = kws + (size_t)bh * L_ * DH;
  const __bf16* vhead = vws + (size_t)bh * DH * L_;

  const int qrow = qblk * 128 + wave * 16 + lm;
  const v16bf qa = frag_bf16(qhead + (size_t)qrow * DH, c);   // scale folded in

  v16bf ones;
#pragma unroll
  for (int i = 0; i < 16; ++i) ones[i] = (__bf16)1.0f;

  float mrow[8];
  v8f o0 = {}, o1 = {}, o2 = {};              // o2 = running row-sum of P
#pragma unroll
  for (int r = 0; r < 8; ++r) mrow[r] = -1e30f;

  __bf16* pw = &pbuf[wave][0];
  const v8f zero = {};

  for (int s0 = 0; s0 < L_; s0 += SBLK) {
    // ---- S = Q @ K^T : four 16-key tiles, K-dim = dh = 32 (one WMMA each) ----
    v8f s[4];
#pragma unroll
    for (int t = 0; t < 4; ++t) {
      v16bf kb = frag_bf16(khead + (size_t)(s0 + t * 16 + lm) * DH, c);
      s[t] = __builtin_amdgcn_wmma_f32_16x16x32_bf16(false, qa, false, kb, (short)0, zero, false, false);
    }

    // ---- V fragments: independent of softmax; issue early to hide latency ----
    v16bf vb0 = frag_bf16(vhead + (size_t)lm * L_ + s0, c);              // keys 0..31, dh 0..15
    v16bf vb1 = frag_bf16(vhead + (size_t)(16 + lm) * L_ + s0, c);       // keys 0..31, dh 16..31
    v16bf vb2 = frag_bf16(vhead + (size_t)lm * L_ + s0 + 32, c);         // keys 32..63, dh 0..15
    v16bf vb3 = frag_bf16(vhead + (size_t)(16 + lm) * L_ + s0 + 32, c);  // keys 32..63, dh 16..31

    // ---- prefetch next key block (speculative; overlaps softmax VALU) ----
    __builtin_prefetch(khead + (size_t)(s0 + SBLK + lm) * DH, 0, 3);
    __builtin_prefetch(vhead + (size_t)lm * L_ + s0 + SBLK, 0, 3);
    __builtin_prefetch(vhead + (size_t)(16 + lm) * L_ + s0 + SBLK, 0, 3);

    // ---- online softmax over 64 new keys (only a MAX reduction needed) ----
#pragma unroll
    for (int r = 0; r < 8; ++r) {
      float t = fmaxf(fmaxf(s[0][r], s[1][r]), fmaxf(s[2][r], s[3][r]));
      t = fmaxf(t, __shfl_xor(t, 1, 32));
      t = fmaxf(t, __shfl_xor(t, 2, 32));
      t = fmaxf(t, __shfl_xor(t, 4, 32));
      t = fmaxf(t, __shfl_xor(t, 8, 32));     // row max over 16-lane half
      const float mn   = fmaxf(mrow[r], t);
      const float corr = __expf(mrow[r] - mn);
      mrow[r] = mn;
      const float p0 = __expf(s[0][r] - mn);
      const float p1 = __expf(s[1][r] - mn);
      const float p2 = __expf(s[2][r] - mn);
      const float p3 = __expf(s[3][r] - mn);
      o0[r] *= corr;
      o1[r] *= corr;
      o2[r] *= corr;                           // row-sum rescales identically
      // scatter P (C layout) into per-wave LDS tile [16 rows x 64 keys]
      __bf16* prow = pw + (r + hi8) * PSTRIDE + lm;
      prow[0]  = (__bf16)p0;
      prow[16] = (__bf16)p1;
      prow[32] = (__bf16)p2;
      prow[48] = (__bf16)p3;
    }

    // ---- gather P as two 16x32 A-fragments from LDS ----
    const __bf16* prdr = pw + lm * PSTRIDE;
    const v16bf pa0 = frag_bf16(prdr, c);        // keys s0    .. s0+31
    const v16bf pa1 = frag_bf16(prdr + 32, c);   // keys s0+32 .. s0+63

    // ---- O += P @ V ; rowsum += P @ 1 ----
    o0 = __builtin_amdgcn_wmma_f32_16x16x32_bf16(false, pa0, false, vb0, (short)0, o0, false, false);
    o1 = __builtin_amdgcn_wmma_f32_16x16x32_bf16(false, pa0, false, vb1, (short)0, o1, false, false);
    o2 = __builtin_amdgcn_wmma_f32_16x16x32_bf16(false, pa0, false, ones, (short)0, o2, false, false);
    o0 = __builtin_amdgcn_wmma_f32_16x16x32_bf16(false, pa1, false, vb2, (short)0, o0, false, false);
    o1 = __builtin_amdgcn_wmma_f32_16x16x32_bf16(false, pa1, false, vb3, (short)0, o1, false, false);
    o2 = __builtin_amdgcn_wmma_f32_16x16x32_bf16(false, pa1, false, ones, (short)0, o2, false, false);
  }

  // ---- normalize and write out [B, L, D] fp32 ----
  const int b = bh / H_;
  const int h = bh % H_;
  const int rowbase = qblk * 128 + wave * 16 + hi8;
#pragma unroll
  for (int r = 0; r < 8; ++r) {
    const float inv = 1.0f / o2[r];            // every lane holds the row sum
    const size_t obase = ((size_t)(b * L_ + rowbase + r)) * D_ + h * DH;
    out[obase + lm]      = o0[r] * inv;
    out[obase + 16 + lm] = o1[r] * inv;
  }
}

// ---------------------------------------------------------------------------
// Launch.  Workspace layout (bf16):
//   [xb 2097152][wq 65536][wk 65536][wv 65536][q 2097152][k 2097152][v 2097152]
// ---------------------------------------------------------------------------
extern "C" void kernel_launch(void* const* d_in, const int* in_sizes, int n_in,
                              void* d_out, int out_size, void* d_ws, size_t ws_size,
                              hipStream_t stream) {
  const float* x  = (const float*)d_in[0];
  const float* Wq = (const float*)d_in[1];
  const float* Wk = (const float*)d_in[2];
  const float* Wv = (const float*)d_in[3];

  const size_t nx = (size_t)M_ * D_;          // 2,097,152
  const size_t nw = (size_t)D_ * D_;          // 65,536
  __bf16* xb  = (__bf16*)d_ws;
  __bf16* wqb = xb  + nx;
  __bf16* wkb = wqb + nw;
  __bf16* wvb = wkb + nw;
  __bf16* qws = wvb + nw;
  __bf16* kws = qws + nx;
  __bf16* vws = kws + nx;

  // Pass 0: bf16 staging (8 elems/thread)
  cvt_bf16_kernel<<<(int)(nx / 8 / 256), 256, 0, stream>>>(x,  xb,  (int)nx);
  cvt_bf16_kernel<<<(int)(nw / 8 / 256), 256, 0, stream>>>(Wq, wqb, (int)nw);
  cvt_bf16_kernel<<<(int)(nw / 8 / 256), 256, 0, stream>>>(Wk, wkb, (int)nw);
  cvt_bf16_kernel<<<(int)(nw / 8 / 256), 256, 0, stream>>>(Wv, wvb, (int)nw);

  // Pass 1: (M/16)*(D/16) = 8192 wave-tiles, 8 waves/block -> 1024 blocks
  qkv_proj_kernel<<<(M_ / 16) * (D_ / 16) / 8, 256, 0, stream>>>(
      xb, wqb, wkb, wvb, qws, kws, vws);

  // Pass 2: B*H*(L/128) = 512 blocks
  attn_kernel<<<B_ * H_ * (L_ / 128), 256, 0, stream>>>(
      qws, kws, vws, (float*)d_out);
}
